// PGCL_31490700214616
// MI455X (gfx1250) — matrix operand order repaired and checked by reading, not compiled, not run on hardware
//
#include <hip/hip_runtime.h>

// ---------------------------------------------------------------------------
// Problem constants (from reference): N=4096 rows, D=512 features, TEMP=0.1
// loss = (1/T^2) * (1/N^2) * [ S(t,rgb) + S(t,depth) + S(t,ir) ],
//   S(a,b) = sum_{ij} (G_a - G_b)_{ij}^2,  G_x = norm(x) @ norm(x)^T
// ---------------------------------------------------------------------------
#define NROWS 4096
#define DDIM  512
#define EPSN  1e-12f

#define AS1 __attribute__((address_space(1)))

typedef _Float16 v16h __attribute__((ext_vector_type(16)));
typedef _Float16 v8h  __attribute__((ext_vector_type(8)));
typedef float    v8f  __attribute__((ext_vector_type(8)));
typedef float    v4f  __attribute__((ext_vector_type(4)));  // builtin vector:
// unlike HIP's float4 class, this is trivially loadable via AS1 pointers.

// Force global (addrspace 1) access paths: global_load_* / global_store_*,
// tracked by LOADcnt/STOREcnt only (FLAT would also tick DScnt and occupy
// the LDS request path).  Int round-trip cast is always-legal inttoptr.
__device__ __forceinline__ const AS1 _Float16* gch(const _Float16* p) {
    return (const AS1 _Float16*)(unsigned long long)p;
}
__device__ __forceinline__ AS1 _Float16* gch_mut(_Float16* p) {
    return (AS1 _Float16*)(unsigned long long)p;
}
__device__ __forceinline__ const AS1 float* gcf(const float* p) {
    return (const AS1 float*)(unsigned long long)p;
}
__device__ __forceinline__ AS1 float* gcf_mut(float* p) {
    return (AS1 float*)(unsigned long long)p;
}

// ---------------------------------------------------------------------------
// Kernel 1: row L2-normalize f32 -> f16.  One wave32 per row (16 elems/lane).
// ---------------------------------------------------------------------------
struct NormArgs {
    const float* src[4];
    _Float16*    dst[4];
};

__global__ __launch_bounds__(256) void pgcl_normalize_kernel(NormArgs args)
{
    const int lane = threadIdx.x & 31;
    const int wave = threadIdx.x >> 5;
    const int row  = blockIdx.x * 8 + wave;   // 0..4095
    const int m    = blockIdx.y;              // 0..3 (t, rgb, depth, ir)

    const AS1 float* s = gcf(args.src[m]) + row * DDIM + lane * 16;

    v4f x[4];
    float ss = 0.0f;
#pragma unroll
    for (int i = 0; i < 4; ++i) {
        x[i] = ((const AS1 v4f*)s)[i];
        ss += x[i][0] * x[i][0] + x[i][1] * x[i][1]
            + x[i][2] * x[i][2] + x[i][3] * x[i][3];
    }
    // wave32 butterfly reduction
#pragma unroll
    for (int off = 16; off > 0; off >>= 1)
        ss += __shfl_xor(ss, off, 32);

    const float scale = 1.0f / fmaxf(sqrtf(ss), EPSN);

    v8h o0, o1;
#pragma unroll
    for (int i = 0; i < 4; ++i) {
        o0[i]     = (_Float16)(x[0][i] * scale);
        o0[i + 4] = (_Float16)(x[1][i] * scale);
        o1[i]     = (_Float16)(x[2][i] * scale);
        o1[i + 4] = (_Float16)(x[3][i] * scale);
    }

    AS1 _Float16* d = gch_mut(args.dst[m]) + row * DDIM + lane * 16;
    *(AS1 v8h*)(d)     = o0;
    *(AS1 v8h*)(d + 8) = o1;
}

// ---------------------------------------------------------------------------
// Kernel 2: fused Gram-difference loss.
//   WG = 256 threads = 8 waves.  Wave (wr,wc) computes a 32x32 block of G
//   at (i0+32*wr, j0+32*wc) via 2x2 v_wmma_f32_16x16x32_f16 tiles, K=512.
//   Pass m=0 keeps the G_t tile in registers; passes m=1..3 accumulate
//   (G_t - G_m)^2 into a per-lane scalar.  One partial sum per workgroup.
// ---------------------------------------------------------------------------
__device__ __forceinline__ v16h pgcl_cat8(v8h lo, v8h hi)
{
    v16h r;
#pragma unroll
    for (int i = 0; i < 8; ++i) { r[i] = lo[i]; r[i + 8] = hi[i]; }
    return r;
}

__device__ __forceinline__ v8h gld8(const AS1 _Float16* p) {
    return *(const AS1 v8h*)p;
}

__global__ __launch_bounds__(256) void pgcl_gram_loss_kernel(
    const _Float16* __restrict__ nt,  const _Float16* __restrict__ na,
    const _Float16* __restrict__ nb,  const _Float16* __restrict__ nc,
    float* __restrict__ partial)
{
    const int lane = threadIdx.x & 31;
    const int wave = threadIdx.x >> 5;
    const int wr   = wave & 1;        // 2 wave-rows
    const int wc   = wave >> 1;       // 4 wave-cols
    const int i0   = blockIdx.x * 64  + wr * 32;
    const int j0   = blockIdx.y * 128 + wc * 32;

    const int rsub  = lane & 15;      // row within 16x16 tile
    const int khalf = lane >> 4;      // 0 or 1 (K half select, ISA 7.12.2)

    const _Float16* mats[4] = { nt, na, nb, nc };

    // 32-bit per-lane row indices off a uniform base -> saddr-form loads
    const int a0i = (i0 + rsub) * DDIM;           // A tile row 0..15
    const int a1i = a0i + 16 * DDIM;              // A tile row 16..31
    const int b0i = (j0 + rsub) * DDIM;           // B tile col 0..15
    const int b1i = b0i + 16 * DDIM;              // B tile col 16..31

    v8f accT[2][2];
    float lsum = 0.0f;

#pragma unroll 1
    for (int m = 0; m < 4; ++m) {
        const AS1 _Float16* p = gch(mats[m]);

        v8f acc[2][2] = {};

#pragma unroll 4
        for (int k = 0; k < DDIM; k += 32) {
            // A fragment (16-bit A 16x32 layout): lane holds row rsub,
            //   halves [k+8*khalf .. +7] and [k+16+8*khalf .. +7]   (2 x b128)
            const int ka = k + khalf * 8;
            // B fragment (B striped across lanes): lane holds column rsub,
            //   halves [k+8*khalf .. +7] and [k+16+8*khalf .. +7]   (2 x b128)
            const int kb = k + khalf * 8;

            v16h a0 = pgcl_cat8(gld8(p + a0i + ka), gld8(p + a0i + ka + 16));
            v16h a1 = pgcl_cat8(gld8(p + a1i + ka), gld8(p + a1i + ka + 16));
            v16h b0 = pgcl_cat8(gld8(p + b0i + kb), gld8(p + b0i + kb + 16));
            v16h b1 = pgcl_cat8(gld8(p + b1i + kb), gld8(p + b1i + kb + 16));

            if (k + 32 < DDIM) {      // prefetch next K-slab (near-temporal)
                __builtin_prefetch(p + a0i + ka + 32, 0, 3);
                __builtin_prefetch(p + b0i + kb + 32, 0, 3);
            }

            acc[0][0] = __builtin_amdgcn_wmma_f32_16x16x32_f16(
                false, a0, false, b0, (short)0, acc[0][0], false, false);
            acc[0][1] = __builtin_amdgcn_wmma_f32_16x16x32_f16(
                false, a0, false, b1, (short)0, acc[0][1], false, false);
            acc[1][0] = __builtin_amdgcn_wmma_f32_16x16x32_f16(
                false, a1, false, b0, (short)0, acc[1][0], false, false);
            acc[1][1] = __builtin_amdgcn_wmma_f32_16x16x32_f16(
                false, a1, false, b1, (short)0, acc[1][1], false, false);
        }

        if (m == 0) {
#pragma unroll
            for (int a = 0; a < 2; ++a)
#pragma unroll
                for (int b = 0; b < 2; ++b) accT[a][b] = acc[a][b];
        } else {
#pragma unroll
            for (int a = 0; a < 2; ++a)
#pragma unroll
                for (int b = 0; b < 2; ++b)
#pragma unroll
                    for (int e = 0; e < 8; ++e) {
                        float d = accT[a][b][e] - acc[a][b][e];
                        lsum += d * d;
                    }
        }
    }

    // wave32 reduce, then cross-wave reduce via LDS; one partial per WG
#pragma unroll
    for (int off = 16; off > 0; off >>= 1)
        lsum += __shfl_xor(lsum, off, 32);

    __shared__ float smem[8];
    if (lane == 0) smem[wave] = lsum;
    __syncthreads();
    if (threadIdx.x == 0) {
        float s = 0.0f;
#pragma unroll
        for (int w = 0; w < 8; ++w) s += smem[w];
        *(gcf_mut(partial) + (blockIdx.y * gridDim.x + blockIdx.x)) = s;
    }
}

// ---------------------------------------------------------------------------
// Kernel 3: deterministic final reduction of 2048 partials + scaling.
// ---------------------------------------------------------------------------
__global__ __launch_bounds__(256) void pgcl_reduce_kernel(
    const float* __restrict__ partial, int n, float* __restrict__ out)
{
    __shared__ float smem[256];
    const AS1 float* gp = gcf(partial);
    float s = 0.0f;
    for (int i = threadIdx.x; i < n; i += 256) s += gp[i];
    smem[threadIdx.x] = s;
    __syncthreads();
#pragma unroll
    for (int off = 128; off > 0; off >>= 1) {
        if (threadIdx.x < off) smem[threadIdx.x] += smem[threadIdx.x + off];
        __syncthreads();
    }
    if (threadIdx.x == 0) {
        // (1/T^2) = 100; each of the 3 losses is a mean over N^2 elements.
        *gcf_mut(out) = smem[0] * (100.0f / ((float)NROWS * (float)NROWS));
    }
}

// ---------------------------------------------------------------------------
// Host launch.  d_in order: fm_rgb, fm_depth, fm_ir, fm_t (all f32 N x D).
// Workspace layout: 4 normalized f16 matrices (4 MiB each) + 2048 partials.
// ---------------------------------------------------------------------------
extern "C" void kernel_launch(void* const* d_in, const int* in_sizes, int n_in,
                              void* d_out, int out_size, void* d_ws, size_t ws_size,
                              hipStream_t stream)
{
    (void)in_sizes; (void)n_in; (void)out_size; (void)ws_size;

    const float* rgb = (const float*)d_in[0];
    const float* dep = (const float*)d_in[1];
    const float* ir  = (const float*)d_in[2];
    const float* t   = (const float*)d_in[3];

    const size_t normBytes = (size_t)NROWS * DDIM * sizeof(_Float16); // 4 MiB
    char* ws = (char*)d_ws;
    _Float16* n_t   = (_Float16*)(ws + 0 * normBytes);
    _Float16* n_rgb = (_Float16*)(ws + 1 * normBytes);
    _Float16* n_dep = (_Float16*)(ws + 2 * normBytes);
    _Float16* n_ir  = (_Float16*)(ws + 3 * normBytes);
    float* partial  = (float*)(ws + 4 * normBytes);   // 2048 floats

    NormArgs na;
    na.src[0] = t;    na.dst[0] = n_t;
    na.src[1] = rgb;  na.dst[1] = n_rgb;
    na.src[2] = dep;  na.dst[2] = n_dep;
    na.src[3] = ir;   na.dst[3] = n_ir;

    // 4096 rows / 8 rows-per-block = 512 blocks, y = matrix index
    pgcl_normalize_kernel<<<dim3(512, 4), 256, 0, stream>>>(na);

    // 4096/64 x 4096/128 = 64 x 32 workgroups
    pgcl_gram_loss_kernel<<<dim3(64, 32), 256, 0, stream>>>(
        n_t, n_rgb, n_dep, n_ir, partial);

    pgcl_reduce_kernel<<<1, 256, 0, stream>>>(partial, 64 * 32, (float*)d_out);
}
